// MoEMLP_47794396070540
// MI455X (gfx1250) — matrix operand order repaired and checked by reading, not compiled, not run on hardware
//
#include <hip/hip_runtime.h>
#include <math.h>

// ---- problem constants (match reference) ----
#define T_TOK 32768
#define D_DIM 768
#define E_EXP 8
#define K_TOP 2
#define I_DIM 768
#define CAP   10240            // ceil(32768*2/8 * 1.25)
#define A_ASN (T_TOK * K_TOP)  // 65536
#define Y_ELEMS ((size_t)T_TOK * D_DIM)

typedef __attribute__((ext_vector_type(16))) __bf16 v16bf;
typedef __attribute__((ext_vector_type(8)))  float  v8f;

union FragBF { v16bf v; uint4 q[2]; };

__device__ __forceinline__ unsigned short f2bf(float f) {
    unsigned int u = __float_as_uint(f);
    u += 0x7fffu + ((u >> 16) & 1u);   // round-to-nearest-even
    return (unsigned short)(u >> 16);
}

// ---- CDNA5 async global->LDS copy (16B per lane), ASYNCcnt tracked ----
__device__ __forceinline__ void async_copy16(unsigned lds_byte_addr, const void* gaddr) {
    asm volatile("global_load_async_to_lds_b128 %0, %1, off"
                 :: "v"(lds_byte_addr),
                    "v"((unsigned long long)(uintptr_t)gaddr)
                 : "memory");
}
__device__ __forceinline__ void wait_async0() {
    asm volatile("s_wait_asynccnt 0x0" ::: "memory");
}

// ---------------- weight prep: f32 [e][R][C] -> bf16 transposed [e][C][R] ----
// grid (C/32, R/32, E), block (32, 8). Coalesced reads and writes via LDS tile.
__global__ __launch_bounds__(256)
void k_transpose_bf16(const float* __restrict__ in, unsigned short* __restrict__ out,
                      int R, int C) {
    __shared__ float tile[32][33];
    int e = blockIdx.z;
    const float* src = in + (size_t)e * R * C;
    unsigned short* dst = out + (size_t)e * R * C;
    int c0 = blockIdx.x * 32, r0 = blockIdx.y * 32;
    int tx = threadIdx.x, ty = threadIdx.y;
#pragma unroll
    for (int i = 0; i < 32; i += 8)
        tile[ty + i][tx] = src[(size_t)(r0 + ty + i) * C + c0 + tx];
    __syncthreads();
#pragma unroll
    for (int i = 0; i < 32; i += 8)
        dst[(size_t)(c0 + ty + i) * R + r0 + tx] = f2bf(tile[tx][ty + i]);
}

__global__ void k_zero_u4(uint4* __restrict__ p, long n4) {
    long i = (long)blockIdx.x * blockDim.x + threadIdx.x;
    long stride = (long)gridDim.x * blockDim.x;
    uint4 z = {0u, 0u, 0u, 0u};
    for (; i < n4; i += stride) p[i] = z;
}

// ---------------- router: logits + top-2 + softmax ----------------
__global__ __launch_bounds__(256)
void k_router(const float* __restrict__ x, const float* __restrict__ router,
              int* __restrict__ e_flat, float* __restrict__ w_flat) {
    int lane = threadIdx.x & 31;
    int wid  = threadIdx.x >> 5;
    int t = blockIdx.x * 8 + wid;
    const float* xr = x + (size_t)t * D_DIM;
    float acc[E_EXP];
#pragma unroll
    for (int e = 0; e < E_EXP; ++e) acc[e] = 0.f;
    for (int i = lane; i < D_DIM; i += 32) {
        float xv = xr[i];
        const float4* rr = (const float4*)(router + (size_t)i * E_EXP);
        float4 r0 = rr[0], r1 = rr[1];
        acc[0] += xv * r0.x; acc[1] += xv * r0.y;
        acc[2] += xv * r0.z; acc[3] += xv * r0.w;
        acc[4] += xv * r1.x; acc[5] += xv * r1.y;
        acc[6] += xv * r1.z; acc[7] += xv * r1.w;
    }
#pragma unroll
    for (int off = 16; off > 0; off >>= 1)
#pragma unroll
        for (int e = 0; e < E_EXP; ++e) acc[e] += __shfl_xor(acc[e], off, 32);
    if (lane == 0) {
        int e0 = 0; float l0 = acc[0];
#pragma unroll
        for (int e = 1; e < E_EXP; ++e) if (acc[e] > l0) { l0 = acc[e]; e0 = e; }
        int e1 = (e0 == 0) ? 1 : 0; float l1 = acc[e1];
#pragma unroll
        for (int e = 0; e < E_EXP; ++e)
            if (e != e0 && acc[e] > l1) { l1 = acc[e]; e1 = e; }
        float w1 = __expf(l1 - l0);
        float s  = 1.f + w1;
        e_flat[2 * t]     = e0;
        e_flat[2 * t + 1] = e1;
        w_flat[2 * t]     = 1.f / s;
        w_flat[2 * t + 1] = w1 / s;
    }
}

// ---------------- ordered capacity scan (single workgroup) ----------------
__global__ __launch_bounds__(1024)
void k_scan(const int* __restrict__ e_flat, int* __restrict__ slot,
            float* __restrict__ out_tail /* counts[8], entropy */) {
    __shared__ int hist[1024][E_EXP];
    __shared__ int totals[E_EXP];
    int t = threadIdx.x;
    int c[E_EXP];
#pragma unroll
    for (int e = 0; e < E_EXP; ++e) c[e] = 0;
    int base = t * 64;
    for (int j = 0; j < 64; ++j) c[e_flat[base + j]]++;
#pragma unroll
    for (int e = 0; e < E_EXP; ++e) hist[t][e] = c[e];
    __syncthreads();
    if (t < E_EXP) {   // serial exclusive scan per expert (tiny)
        int run = 0;
        for (int i = 0; i < 1024; ++i) { int v = hist[i][t]; hist[i][t] = run; run += v; }
        totals[t] = run;
    }
    __syncthreads();
#pragma unroll
    for (int e = 0; e < E_EXP; ++e) c[e] = hist[t][e];
    for (int j = 0; j < 64; ++j) {
        int a = base + j;
        int e = e_flat[a];
        int pos = c[e]++;
        slot[a] = (pos < CAP) ? (e * CAP + pos) : -1;
    }
    if (t == 0) {
        float ent = 0.f;
        for (int e = 0; e < E_EXP; ++e) {
            float cnt = (float)totals[e];
            out_tail[e] = cnt;
            float ld = cnt / (float)A_ASN;
            ent -= ld * logf(ld + 1e-6f);
        }
        out_tail[E_EXP] = ent;
    }
}

// ---------------- dispatch: scatter x rows (f32 -> bf16) ----------------
__global__ __launch_bounds__(256)
void k_dispatch(const float* __restrict__ x, const int* __restrict__ slot,
                unsigned short* __restrict__ xb) {
    int lane = threadIdx.x & 31;
    int wid  = threadIdx.x >> 5;
    int a = blockIdx.x * 8 + wid;
    int s = slot[a];
    if (s < 0) return;
    int tok = a >> 1;
    const float* src = x + (size_t)tok * D_DIM;
    unsigned short* dst = xb + (size_t)s * D_DIM;
    for (int i = lane; i < D_DIM; i += 32) dst[i] = f2bf(src[i]);
}

// ---------------- GEMM1: x_buf @ w_up_gate + SwiGLU -> act (bf16) ----------
// grid (I/64, CAP/128, E), block 256 (8 waves). Tile: 128 rows x (64 gate + 64 up).
// Fully async-fed, double-buffered LDS. Weights pre-transposed: wugT[e][n(1536)][k(768)].
__global__ __launch_bounds__(256)
void k_gemm_up_swiglu(const unsigned short* __restrict__ xb,
                      const unsigned short* __restrict__ wugT,
                      unsigned short* __restrict__ actb) {
    __shared__ __align__(16) unsigned short lds_a[2][128 * 32];   // [row][k]
    __shared__ __align__(16) unsigned short lds_bg[2][64 * 32];   // [n][k]
    __shared__ __align__(16) unsigned short lds_bu[2][64 * 32];   // [n][k]
    int e  = blockIdx.z;
    int m0 = blockIdx.y * 128;
    int n0 = blockIdx.x * 64;
    int tid  = threadIdx.x;
    int lane = tid & 31, w = tid >> 5;
    int wm = w & 3, wn = w >> 2;                 // wave grid: 4 (M) x 2 (N)
    const unsigned short* Ae  = xb   + (size_t)e * CAP * D_DIM;
    const unsigned short* WgT = wugT + (size_t)e * (2 * I_DIM) * D_DIM;
    v8f zero8 = {0.f,0.f,0.f,0.f,0.f,0.f,0.f,0.f};
    v8f cg[2][2], cu[2][2];
#pragma unroll
    for (int mi = 0; mi < 2; ++mi)
#pragma unroll
        for (int ni = 0; ni < 2; ++ni) { cg[mi][ni] = zero8; cu[mi][ni] = zero8; }

    auto loadA = [&](int buf, int k0) {
#pragma unroll
        for (int p = 0; p < 2; ++p) {
            int idx = tid + p * 256;            // 0..511 chunks of 16B
            int row = idx >> 2, c8 = idx & 3;
            unsigned dst = (unsigned)(uintptr_t)&lds_a[buf][row * 32 + c8 * 8];
            async_copy16(dst, Ae + (size_t)(m0 + row) * D_DIM + k0 + c8 * 8);
        }
    };
    auto loadB = [&](int buf, int k0) {         // 256 chunks per sub-tile
        int row = tid >> 2, c8 = tid & 3;       // row 0..63, c8 0..3
        unsigned dg = (unsigned)(uintptr_t)&lds_bg[buf][row * 32 + c8 * 8];
        async_copy16(dg, WgT + (size_t)(n0 + row) * D_DIM + k0 + c8 * 8);
        unsigned du = (unsigned)(uintptr_t)&lds_bu[buf][row * 32 + c8 * 8];
        async_copy16(du, WgT + (size_t)(I_DIM + n0 + row) * D_DIM + k0 + c8 * 8);
    };

    loadA(0, 0); loadB(0, 0);
    for (int k0 = 0, it = 0; k0 < D_DIM; k0 += 32, ++it) {
        int cur = it & 1;
        wait_async0();          // my async writes to lds done
        __syncthreads();        // everyone's writes visible, prev reads done
        if (k0 + 32 < D_DIM) { loadA(cur ^ 1, k0 + 32); loadB(cur ^ 1, k0 + 32); }

        int kbA = (lane < 16) ? 0 : 8;          // A: k chunks {kbA..+7, kbA+16..+23}
        int kbB = (lane < 16) ? 0 : 8;          // B: k chunks {kbB..+7, kbB+16..+23}
        FragBF a[2], bg[2], bu[2];
#pragma unroll
        for (int mi = 0; mi < 2; ++mi) {
            int row = wm * 32 + mi * 16 + (lane & 15);
            a[mi].q[0] = *(const uint4*)(&lds_a[cur][row * 32 + kbA]);
            a[mi].q[1] = *(const uint4*)(&lds_a[cur][row * 32 + kbA + 16]);
        }
#pragma unroll
        for (int ni = 0; ni < 2; ++ni) {
            int n = wn * 32 + ni * 16 + (lane & 15);
            bg[ni].q[0] = *(const uint4*)(&lds_bg[cur][n * 32 + kbB]);
            bg[ni].q[1] = *(const uint4*)(&lds_bg[cur][n * 32 + kbB + 16]);
            bu[ni].q[0] = *(const uint4*)(&lds_bu[cur][n * 32 + kbB]);
            bu[ni].q[1] = *(const uint4*)(&lds_bu[cur][n * 32 + kbB + 16]);
        }
#pragma unroll
        for (int mi = 0; mi < 2; ++mi)
#pragma unroll
            for (int ni = 0; ni < 2; ++ni) {
                cg[mi][ni] = __builtin_amdgcn_wmma_f32_16x16x32_bf16(
                    false, a[mi].v, false, bg[ni].v, (short)0, cg[mi][ni], false, false);
                cu[mi][ni] = __builtin_amdgcn_wmma_f32_16x16x32_bf16(
                    false, a[mi].v, false, bu[ni].v, (short)0, cu[mi][ni], false, false);
            }
    }
    // epilogue: silu(gate)*up -> bf16 (fast sigmoid: v_exp + v_rcp)
    int mhi = (lane < 16) ? 0 : 8;
#pragma unroll
    for (int mi = 0; mi < 2; ++mi)
#pragma unroll
        for (int ni = 0; ni < 2; ++ni) {
            int nb = n0 + wn * 32 + ni * 16 + (lane & 15);
#pragma unroll
            for (int r = 0; r < 8; ++r) {
                int m = m0 + wm * 32 + mi * 16 + r + mhi;
                float g = cg[mi][ni][r];
                float u = cu[mi][ni][r];
                float sig = __builtin_amdgcn_rcpf(1.f + __expf(-g));
                actb[((size_t)e * CAP + m) * I_DIM + nb] = f2bf(g * sig * u);
            }
        }
}

// ---------------- GEMM2: act @ w_down -> y_buf (f32) ----------------
// grid (D/128, CAP/128, E), block 256 (8 waves). Tile 128x128, wave 64x32.
// Fully async-fed, double-buffered LDS. Weights pre-transposed: wdnT[e][n(768)][k(768)].
__global__ __launch_bounds__(256)
void k_gemm_down(const unsigned short* __restrict__ actb,
                 const unsigned short* __restrict__ wdnT,
                 float* __restrict__ yb) {
    __shared__ __align__(16) unsigned short lds_a[2][128 * 32];   // [row][k]
    __shared__ __align__(16) unsigned short lds_b[2][128 * 32];   // [n][k]
    int e  = blockIdx.z;
    int m0 = blockIdx.y * 128;
    int n0 = blockIdx.x * 128;
    int tid  = threadIdx.x;
    int lane = tid & 31, w = tid >> 5;
    int wm = w & 1, wn = w >> 1;                 // wave grid: 2 (M) x 4 (N)
    const unsigned short* Ae  = actb + (size_t)e * CAP * I_DIM;
    const unsigned short* WeT = wdnT + (size_t)e * D_DIM * I_DIM;
    v8f zero8 = {0.f,0.f,0.f,0.f,0.f,0.f,0.f,0.f};
    v8f c[4][2];
#pragma unroll
    for (int mi = 0; mi < 4; ++mi)
#pragma unroll
        for (int ni = 0; ni < 2; ++ni) c[mi][ni] = zero8;

    auto loadA = [&](int buf, int k0) {
#pragma unroll
        for (int p = 0; p < 2; ++p) {
            int idx = tid + p * 256;
            int row = idx >> 2, c8 = idx & 3;
            unsigned dst = (unsigned)(uintptr_t)&lds_a[buf][row * 32 + c8 * 8];
            async_copy16(dst, Ae + (size_t)(m0 + row) * I_DIM + k0 + c8 * 8);
        }
    };
    auto loadB = [&](int buf, int k0) {
#pragma unroll
        for (int p = 0; p < 2; ++p) {
            int idx = tid + p * 256;
            int row = idx >> 2, c8 = idx & 3;   // row 0..127
            unsigned dst = (unsigned)(uintptr_t)&lds_b[buf][row * 32 + c8 * 8];
            async_copy16(dst, WeT + (size_t)(n0 + row) * I_DIM + k0 + c8 * 8);
        }
    };

    loadA(0, 0); loadB(0, 0);
    for (int k0 = 0, it = 0; k0 < I_DIM; k0 += 32, ++it) {
        int cur = it & 1;
        wait_async0();
        __syncthreads();
        if (k0 + 32 < I_DIM) { loadA(cur ^ 1, k0 + 32); loadB(cur ^ 1, k0 + 32); }

        int kbA = (lane < 16) ? 0 : 8;
        int kbB = (lane < 16) ? 0 : 8;
        FragBF a[4], b[2];
#pragma unroll
        for (int mi = 0; mi < 4; ++mi) {
            int row = wm * 64 + mi * 16 + (lane & 15);
            a[mi].q[0] = *(const uint4*)(&lds_a[cur][row * 32 + kbA]);
            a[mi].q[1] = *(const uint4*)(&lds_a[cur][row * 32 + kbA + 16]);
        }
#pragma unroll
        for (int ni = 0; ni < 2; ++ni) {
            int n = wn * 32 + ni * 16 + (lane & 15);
            b[ni].q[0] = *(const uint4*)(&lds_b[cur][n * 32 + kbB]);
            b[ni].q[1] = *(const uint4*)(&lds_b[cur][n * 32 + kbB + 16]);
        }
#pragma unroll
        for (int mi = 0; mi < 4; ++mi)
#pragma unroll
            for (int ni = 0; ni < 2; ++ni)
                c[mi][ni] = __builtin_amdgcn_wmma_f32_16x16x32_bf16(
                    false, a[mi].v, false, b[ni].v, (short)0, c[mi][ni], false, false);
    }
    int mhi = (lane < 16) ? 0 : 8;
#pragma unroll
    for (int mi = 0; mi < 4; ++mi)
#pragma unroll
        for (int ni = 0; ni < 2; ++ni) {
            int nb = n0 + wn * 32 + ni * 16 + (lane & 15);
#pragma unroll
            for (int r = 0; r < 8; ++r) {
                int m = m0 + wm * 64 + mi * 16 + r + mhi;
                yb[((size_t)e * CAP + m) * D_DIM + nb] = c[mi][ni][r];
            }
        }
}

// ---------------- combine: y[t] = w0*yb[s0] + w1*yb[s1] ----------------
__global__ __launch_bounds__(256)
void k_combine(const float* __restrict__ yb, const int* __restrict__ slot,
               const float* __restrict__ w_flat, float* __restrict__ y) {
    int lane = threadIdx.x & 31;
    int wid  = threadIdx.x >> 5;
    int t = blockIdx.x * 8 + wid;
    int s0 = slot[2 * t], s1 = slot[2 * t + 1];
    float w0 = w_flat[2 * t], w1 = w_flat[2 * t + 1];
    const float4* p0 = (const float4*)(yb + (s0 >= 0 ? (size_t)s0 * D_DIM : 0));
    const float4* p1 = (const float4*)(yb + (s1 >= 0 ? (size_t)s1 * D_DIM : 0));
    float4* out = (float4*)(y + (size_t)t * D_DIM);
#pragma unroll
    for (int i = 0; i < 6; ++i) {
        int j = i * 32 + lane;
        float4 acc = {0.f, 0.f, 0.f, 0.f};
        if (s0 >= 0) {
            float4 v = p0[j];
            acc.x += w0 * v.x; acc.y += w0 * v.y; acc.z += w0 * v.z; acc.w += w0 * v.w;
        }
        if (s1 >= 0) {
            float4 v = p1[j];
            acc.x += w1 * v.x; acc.y += w1 * v.y; acc.z += w1 * v.z; acc.w += w1 * v.w;
        }
        out[j] = acc;
    }
}

// ---------------- launcher ----------------
extern "C" void kernel_launch(void* const* d_in, const int* in_sizes, int n_in,
                              void* d_out, int out_size, void* d_ws, size_t ws_size,
                              hipStream_t stream) {
    const float* x      = (const float*)d_in[0];   // (B,S,D) = (T, 768)
    const float* router = (const float*)d_in[1];   // (768, 8)
    const float* wug    = (const float*)d_in[2];   // (8, 768, 1536)
    const float* wdn    = (const float*)d_in[3];   // (8, 768, 768)
    float* y = (float*)d_out;                      // y | counts[8] | entropy

    const long n_wug = (long)E_EXP * D_DIM * 2 * I_DIM;   // 9,437,184
    const long n_wdn = (long)E_EXP * I_DIM * D_DIM;       // 4,718,592
    const long n_xbuf = (long)E_EXP * CAP * D_DIM;        // 62,914,560 bf16

    char* ws = (char*)d_ws;
    size_t off = 0;
    unsigned short* wug_bfT = (unsigned short*)(ws + off); off += (size_t)n_wug * 2;
    unsigned short* wdn_bfT = (unsigned short*)(ws + off); off += (size_t)n_wdn * 2;
    int*   e_flat = (int*)(ws + off);   off += (size_t)A_ASN * 4;
    float* w_flat = (float*)(ws + off); off += (size_t)A_ASN * 4;
    int*   slot   = (int*)(ws + off);   off += (size_t)A_ASN * 4;
    unsigned short* x_buf   = (unsigned short*)(ws + off); off += (size_t)n_xbuf * 2;
    unsigned short* act_buf = (unsigned short*)(ws + off); off += (size_t)n_xbuf * 2;
    float* y_buf = (float*)(ws + off);  off += (size_t)n_xbuf * 4;
    (void)in_sizes; (void)n_in; (void)out_size; (void)ws_size;

    // weights: convert + transpose to [e][n][k] bf16 (async-copy friendly)
    k_transpose_bf16<<<dim3(2 * I_DIM / 32, D_DIM / 32, E_EXP), dim3(32, 8), 0, stream>>>(
        wug, wug_bfT, D_DIM, 2 * I_DIM);
    k_transpose_bf16<<<dim3(D_DIM / 32, I_DIM / 32, E_EXP), dim3(32, 8), 0, stream>>>(
        wdn, wdn_bfT, I_DIM, D_DIM);
    k_router<<<T_TOK / 8, 256, 0, stream>>>(x, router, e_flat, w_flat);
    k_scan<<<1, 1024, 0, stream>>>(e_flat, slot, y + Y_ELEMS);
    k_zero_u4<<<4096, 256, 0, stream>>>((uint4*)x_buf, n_xbuf / 8);
    k_dispatch<<<A_ASN / 8, 256, 0, stream>>>(x, slot, x_buf);
    k_gemm_up_swiglu<<<dim3(I_DIM / 64, CAP / 128, E_EXP), 256, 0, stream>>>(
        x_buf, wug_bfT, act_buf);
    k_gemm_down<<<dim3(D_DIM / 128, CAP / 128, E_EXP), 256, 0, stream>>>(
        act_buf, wdn_bfT, y_buf);
    k_combine<<<T_TOK / 8, 256, 0, stream>>>(y_buf, slot, w_flat, y);
}
